// ExportableStudentSNN1d_70669391888495
// MI455X (gfx1250) — matrix operand order repaired and checked by reading, not compiled, not run on hardware
//

#include <hip/hip_runtime.h>
#include <hip/hip_bf16.h>
#include <stddef.h>

// ---------------------------------------------------------------------------
// SNN pipeline for MI455X (gfx1250, wave32, WMMA).
//   Layer1 (TAU1=1.0 -> stateless): s1 = ((conv1(x_t)+b1)*GAIN >= TH1)
//   Layer2: WMMA f16 GEMM (im2col-in-LDS) + fused LIF(v2, TAU2=0.9) + spike count
//   Readout: spike counts (b,c2) -> /(T*L) -> FC 256x4
// ---------------------------------------------------------------------------

#define B_    64
#define CIN_  12
#define C1_   128
#define C2_   256
#define K_    9
#define L_    2048
#define T_    20
#define KK_   (C1_ * K_)       // 1152, reordered as kk = k*128 + ci
#define NKS_  (KK_ / 32)       // 36 WMMA K-steps

#define LT_   64               // l-tile per conv2 block
#define MT_   (LT_ / 16)       // 4 m-tiles per wave
#define ROWS_ (LT_ + 8)        // LDS rows incl. halo

static constexpr float GAIN_  = 3.0f;
static constexpr float ITAU2_ = 1.0f / 0.9f;
static constexpr float TH1_   = 0.02f;
static constexpr float TH2_   = 0.02f;

typedef __attribute__((ext_vector_type(16))) _Float16 v16h;
typedef __attribute__((ext_vector_type(8)))  _Float16 v8h;
typedef __attribute__((ext_vector_type(8)))  float    v8f;

// ---------------- workspace layout (bytes) ----------------
#define OFF_V2   ((size_t)0)                          // B*C2*L f32  = 128 MB
#define OFF_S1   ((size_t)134217728)                  // B*L*C1 f16  =  32 MB
#define OFF_W2Q  ((size_t)(134217728 + 33554432))     // KK*C2 f16   = 576 KB
#define OFF_SACC ((size_t)(134217728 + 33554432 + 589824)) // B*C2 f32 = 64 KB
// NOTE: SACC sits directly after W2Q, so the unconditional +64B weight
// prefetch in the K-loop always targets valid workspace memory.

// ---------------------------------------------------------------------------
__global__ void zero_f32_kernel(float* __restrict__ p, long n) {
    long i = (long)blockIdx.x * blockDim.x + threadIdx.x;
    long stride = (long)gridDim.x * blockDim.x;
    for (; i < n; i += stride) p[i] = 0.0f;
}

// W2 [C2][C1][K] f32  ->  W2q [C2][kk = k*128+ci] f16  (B-operand, kk-contiguous)
__global__ void pack_w2_kernel(const float* __restrict__ W2, _Float16* __restrict__ W2q) {
    int i = blockIdx.x * 256 + threadIdx.x;          // i = kk*256 + c2
    if (i >= KK_ * C2_) return;
    int c2 = i & 255;
    int kk = i >> 8;
    int ci = kk & 127;
    int k  = kk >> 7;
    W2q[(size_t)c2 * KK_ + kk] = (_Float16)W2[(c2 * C1_ + ci) * K_ + k];
}

// ---------------------------------------------------------------------------
// conv1 + threshold:  s1[b][l][ci_out] f16 in {0,1}
// grid (L/128, B), block 128 (one thread per l position)
// ---------------------------------------------------------------------------
__global__ void conv1_lif1_kernel(const float* __restrict__ x,
                                  const float* __restrict__ W1,
                                  const float* __restrict__ b1,
                                  _Float16* __restrict__ s1, int t) {
    __shared__ float w1s[C1_ * CIN_ * K_];   // 55296 B
    __shared__ float b1s[C1_];
    __shared__ float xs[CIN_ * 136];         // l-tile + halo(8)

    const int tid = threadIdx.x;
    const int l0  = blockIdx.x * 128;
    const int b   = blockIdx.y;

    for (int i = tid; i < C1_ * CIN_ * K_; i += 128) w1s[i] = W1[i];
    b1s[tid] = b1[tid];
    for (int i = tid; i < CIN_ * 136; i += 128) {
        int ci = i / 136, p = i % 136;
        int gl = l0 - 4 + p;
        float v = 0.0f;
        if (gl >= 0 && gl < L_)
            v = x[(((size_t)b * CIN_ + ci) * L_ + gl) * T_ + t];
        xs[i] = v;
    }
    __syncthreads();

    const int l = l0 + tid;
    const size_t srow = ((size_t)b * L_ + l) * C1_;

    for (int cb = 0; cb < C1_; cb += 8) {
        float acc[8];
#pragma unroll
        for (int j = 0; j < 8; ++j) acc[j] = b1s[cb + j];
        for (int ci = 0; ci < CIN_; ++ci) {
#pragma unroll
            for (int k = 0; k < K_; ++k) {
                float xv = xs[ci * 136 + tid + k];       // same 128B line across lanes
#pragma unroll
                for (int j = 0; j < 8; ++j)              // w1s: LDS broadcast (uniform)
                    acc[j] += w1s[(cb + j) * (CIN_ * K_) + ci * K_ + k] * xv;
            }
        }
        __align__(16) _Float16 h[8];
#pragma unroll
        for (int j = 0; j < 8; ++j)
            h[j] = (acc[j] * GAIN_ >= TH1_) ? (_Float16)1.0f : (_Float16)0.0f;
        *(float4*)(s1 + srow + cb) = *(const float4*)h;  // 16B coalesced store
    }
}

// ---------------------------------------------------------------------------
// conv2 as WMMA f16 GEMM + fused LIF2 + spike-count accumulation.
//   A (16x32 f16)  = spike tile, M = l, K = kk chunk      (from LDS)
//   B (32x16 f16)  = packed weights W2q, N = c2           (from global/L2)
//   D (16x16 f32)  = pre-activation; epilogue: v2 update, spike, atomic count
// grid (L/64, B), block 256 = 8 waves.
// Wave w owns a 4(m=l)x2(n=c2) register tile: 8 WMMA per K-step against
// 8 ds_load_b128 + 4 global_load_b128.
// ---------------------------------------------------------------------------
__global__ void conv2_lif2_kernel(const _Float16* __restrict__ s1,
                                  const _Float16* __restrict__ W2q,
                                  const float* __restrict__ b2,
                                  float* __restrict__ v2,
                                  float* __restrict__ spike_acc) {
    __shared__ _Float16 lds[ROWS_ * C1_];   // rows: l0-4 .. l0+LT_+3 (18 KB)

    const int tid  = threadIdx.x;
    const int l0   = blockIdx.x * LT_;
    const int b    = blockIdx.y;
    const int wv   = tid >> 5;
    const int lane = tid & 31;
    const int mrow  = lane & 15;
    const int mhalf = lane >> 4;

    // ---- stage spike tile (with halo) into LDS as u32 copies (coalesced) ----
    const unsigned* s1u = (const unsigned*)(s1 + (size_t)b * L_ * C1_);
    unsigned* ldsu = (unsigned*)lds;
    for (int i = tid; i < ROWS_ * (C1_ / 2); i += 256) {
        int r = i >> 6;                 // / 64 u32 per row
        int c = i & 63;
        int gl = l0 - 4 + r;
        unsigned v = 0u;
        if (gl >= 0 && gl < L_) v = s1u[gl * (C1_ / 2) + c];
        ldsu[i] = v;
    }
    __syncthreads();

    v8f acc[MT_][2] = {};   // [l-tile][c2-tile]

    for (int ks = 0; ks < NKS_; ++ks) {
        const int k  = ks >> 2;          // conv tap
        const int c0 = (ks & 3) << 5;    // ci base of 32-chunk

        // A fragments (spikes). ISA lane layout: lane&15 = M row; per-lane K set
        // is {0-7,16-23}+8*half -> assemble from two 16B LDS loads.
        v16h A[MT_];
#pragma unroll
        for (int mi = 0; mi < MT_; ++mi) {
            const _Float16* base = &lds[(mi * 16 + mrow + k) * C1_ + c0];
            union { v16h v; v8h h[2]; } u;
            u.h[0] = *(const v8h*)(base + mhalf * 8);
            u.h[1] = *(const v8h*)(base + 16 + mhalf * 8);
            A[mi] = u.v;
        }

        // B fragments (weights): lane&15 = N col (c2), K set {0-7,16-23}+8*half (two 16B loads).
        // Prefetch is unconditional (avoids last-iteration peel -> spills);
        // past-the-end addresses land in workspace / are speculative-dropped.
        v16h Bf[2];
#pragma unroll
        for (int ni = 0; ni < 2; ++ni) {
            const int c2row = (2 * wv + ni) * 16 + mrow;
            const _Float16* wp = W2q + (size_t)c2row * KK_ + ks * 32;
            union { v16h v; v8h h[2]; } ub;
            ub.h[0] = *(const v8h*)(wp + mhalf * 8);
            ub.h[1] = *(const v8h*)(wp + 16 + mhalf * 8);
            Bf[ni] = ub.v;
            __builtin_prefetch(wp + 32, 0, 1);   // global_prefetch_b8
        }

#pragma unroll
        for (int mi = 0; mi < MT_; ++mi)
#pragma unroll
            for (int ni = 0; ni < 2; ++ni)
                acc[mi][ni] = __builtin_amdgcn_wmma_f32_16x16x32_f16(
                    false, A[mi], false, Bf[ni], (short)0, acc[mi][ni], false, false);
    }

    // ---- fused epilogue: bias+gain, LIF2 state update, spike count ----
#pragma unroll
    for (int ni = 0; ni < 2; ++ni) {
        const int c2 = (2 * wv + ni) * 16 + mrow;   // lane-resident c2
        const float bb = b2[c2];
        const size_t vbase = ((size_t)b * C2_ + c2) * L_;
        float scnt = 0.0f;
#pragma unroll
        for (int mi = 0; mi < MT_; ++mi) {
            union { v8f v; float f[8]; } u;
            u.v = acc[mi][ni];
#pragma unroll
            for (int e = 0; e < 8; ++e) {
                const int l = l0 + mi * 16 + mhalf * 8 + e;  // C layout: M = e (+8 hi half)
                const float a2 = (u.f[e] + bb) * GAIN_;
                float v = v2[vbase + l];
                v += (a2 - v) * ITAU2_;
                const float sp = (v >= TH2_) ? 1.0f : 0.0f;
                v2[vbase + l] = (sp > 0.0f) ? 0.0f : v;
                scnt += sp;
            }
        }
        atomicAdd(&spike_acc[b * C2_ + c2], scnt);   // L2 atomic (1 per lane per n-tile)
    }
}

// ---------------------------------------------------------------------------
// readout: out[b][cls] = bfc[cls] + (1/(T*L)) * sum_c2 acc[b,c2]*Wfc[cls,c2]
// ---------------------------------------------------------------------------
__global__ void fc_kernel(const float* __restrict__ spike_acc,
                          const float* __restrict__ Wfc,
                          const float* __restrict__ bfc,
                          float* __restrict__ out) {
    __shared__ float red[256];
    const int b = blockIdx.x, tid = threadIdx.x;
    const float sv = spike_acc[b * C2_ + tid] * (1.0f / ((float)T_ * (float)L_));
    for (int cls = 0; cls < 4; ++cls) {
        red[tid] = sv * Wfc[cls * C2_ + tid];
        __syncthreads();
        for (int s = 128; s > 0; s >>= 1) {
            if (tid < s) red[tid] += red[tid + s];
            __syncthreads();
        }
        if (tid == 0) out[b * 4 + cls] = red[0] + bfc[cls];
        __syncthreads();
    }
}

// ---------------------------------------------------------------------------
extern "C" void kernel_launch(void* const* d_in, const int* in_sizes, int n_in,
                              void* d_out, int out_size, void* d_ws, size_t ws_size,
                              hipStream_t stream) {
    (void)in_sizes; (void)n_in; (void)out_size; (void)ws_size;

    const float* x   = (const float*)d_in[0];
    const float* W1  = (const float*)d_in[1];
    const float* b1  = (const float*)d_in[2];
    const float* W2  = (const float*)d_in[3];
    const float* b2  = (const float*)d_in[4];
    const float* Wfc = (const float*)d_in[5];
    const float* bfc = (const float*)d_in[6];
    float* out = (float*)d_out;

    char* ws = (char*)d_ws;
    float*     v2   = (float*)(ws + OFF_V2);
    _Float16*  s1   = (_Float16*)(ws + OFF_S1);
    _Float16*  W2q  = (_Float16*)(ws + OFF_W2Q);
    float*     sacc = (float*)(ws + OFF_SACC);

    // state init (every call: deterministic, graph-replay safe)
    zero_f32_kernel<<<4096, 256, 0, stream>>>(v2, (long)B_ * C2_ * L_);
    zero_f32_kernel<<<64,   256, 0, stream>>>(sacc, (long)B_ * C2_);
    pack_w2_kernel<<<(KK_ * C2_ + 255) / 256, 256, 0, stream>>>(W2, W2q);

    dim3 g1(L_ / 128, B_);
    dim3 g2(L_ / LT_, B_);
    for (int t = 0; t < T_; ++t) {
        conv1_lif1_kernel<<<g1, 128, 0, stream>>>(x, W1, b1, s1, t);
        conv2_lif2_kernel<<<g2, 256, 0, stream>>>(s1, W2q, b2, v2, sacc);
    }

    fc_kernel<<<B_, 256, 0, stream>>>(sacc, Wfc, bfc, out);
}
